// AgentAttention_55825984914155
// MI455X (gfx1250) — matrix operand order repaired and checked by reading, not compiled, not run on hardware
//
#include <hip/hip_runtime.h>

typedef __attribute__((ext_vector_type(8)))  __bf16 v8bf;
typedef __attribute__((ext_vector_type(16))) __bf16 v16bf;
typedef __attribute__((ext_vector_type(8)))  float  v8f;

#define B_     16
#define N_     3136
#define C_     384
#define H_     56
#define HEADS_ 8
#define DH_    48
#define AG_    49
#define M_     (B_*N_)                 // 50176
#define SCALE_ 0.14433756729740643f    // 48^-0.5

// ---------------- WMMA helpers (gfx1250 fragment layouts, ISA 7.12.2) ----------------

// 16-bit A fragment (16x32): per lane two contiguous 8-element groups:
// lanes 0-15: K {0..7, 16..23}; lanes 16-31: K {8..15, 24..31}. Caller passes
// base + (lane&16 ? 8 : 0); we read [0..7] and [16..23].
__device__ __forceinline__ v16bf afrag(const __bf16* p) {
  v8bf lo = *(const v8bf*)p;
  v8bf hi = *(const v8bf*)(p + 16);
  return __builtin_shufflevector(lo, hi, 0,1,2,3,4,5,6,7,8,9,10,11,12,13,14,15);
}

__device__ __forceinline__ v8f wmma_bf(v16bf a, v16bf b, v8f c) {
  return __builtin_amdgcn_wmma_f32_16x16x32_bf16(false, a, false, b, (short)0, c, false, false);
}

// ---------------- small prep kernels ----------------

__global__ void f32_to_bf16_kernel(const float* __restrict__ src, __bf16* __restrict__ dst, long n) {
  long i = (long)blockIdx.x*256 + threadIdx.x;
  if (i < n) dst[i] = (__bf16)src[i];
}

// src: K(=384) x Ncols row-major f32  ->  dst: Ncols x K row-major bf16 (transposed)
__global__ void transpose_bf16_kernel(const float* __restrict__ src, __bf16* __restrict__ dst, int Ncols) {
  int idx = blockIdx.x*256 + threadIdx.x;
  if (idx >= Ncols*C_) return;
  int k = idx % C_;
  int n = idx / C_;
  dst[idx] = (__bf16)src[(size_t)k*Ncols + n];
}

// pos_bias[h][a][n] = an[h][a][n] + ah[h][a][row] + aw[h][a][col]
__global__ void pos_bias_kernel(const float* __restrict__ an, const float* __restrict__ ahb,
                                const float* __restrict__ awb, float* __restrict__ out) {
  int idx = blockIdx.x*256 + threadIdx.x;
  if (idx >= HEADS_*AG_*N_) return;
  int n  = idx % N_;
  int ha = idx / N_;
  int row = n / H_, col = n % H_;
  out[idx] = an[idx] + ahb[ha*H_ + row] + awb[ha*H_ + col];
}

// agent_bias[h][n][a] = na[h][a][n] + ha[h][row][a] + wa[h][col][a]
__global__ void agent_bias_kernel(const float* __restrict__ na, const float* __restrict__ hab,
                                  const float* __restrict__ wab, float* __restrict__ out) {
  int idx = blockIdx.x*256 + threadIdx.x;
  if (idx >= HEADS_*N_*AG_) return;
  int a   = idx % AG_;
  int rem = idx / AG_;
  int n   = rem % N_;
  int h   = rem / N_;
  int row = n / H_, col = n % H_;
  out[idx] = na[((size_t)(h*AG_ + a))*N_ + n] + hab[(h*H_ + row)*AG_ + a] + wab[(h*H_ + col)*AG_ + a];
}

// agent[b][a][c] = mean over 8x8 block of q
__global__ void agent_pool_kernel(const __bf16* __restrict__ q_bf, __bf16* __restrict__ agent_bf) {
  int idx = blockIdx.x*256 + threadIdx.x;
  if (idx >= B_*AG_*C_) return;
  int cc  = idx % C_;
  int rem = idx / C_;
  int a   = rem % AG_;
  int b   = rem / AG_;
  int p1 = a / 7, p2 = a % 7;
  float s = 0.f;
  for (int i = 0; i < 8; ++i)
    for (int j = 0; j < 8; ++j) {
      int n = (p1*8 + i)*H_ + (p2*8 + j);
      s += (float)q_bf[((size_t)(b*N_ + n))*C_ + cc];
    }
  agent_bf[idx] = (__bf16)(s * (1.f/64.f));
}

// ---------------- main GEMM: C[M x Ncols] = A[M x 384] * W, W given as Bt[Ncols x 384] ----------------
// 8 waves/block, each wave owns a 32x32 tile -> block tile 128M x 64N. 12 K-steps, 4 wmma each.
__global__ __launch_bounds__(256) void gemm_bf16_kernel(const __bf16* __restrict__ A,
                                                        const __bf16* __restrict__ Bt,
                                                        __bf16* __restrict__ Cc, int Ncols) {
  const int K = C_;
  const int lane = threadIdx.x & 31;
  const int wave = threadIdx.x >> 5;
  const int wm = wave >> 1, wn = wave & 1;
  const size_t m0 = (size_t)blockIdx.x*128 + (size_t)wm*32;
  const size_t n0 = (size_t)blockIdx.y*64  + (size_t)wn*32;
  const int aoff = (lane & 16) ? 8  : 0;
  const int boff = (lane & 16) ? 8  : 0;
  const __bf16* a0p = A  + (m0 + (lane & 15))*K + aoff;
  const __bf16* a1p = a0p + (size_t)16*K;
  const __bf16* b0p = Bt + (n0 + (lane & 15))*K + boff;
  const __bf16* b1p = b0p + (size_t)16*K;
  // warm the far halves of this block's working set while the first K-steps issue
  __builtin_prefetch(a0p + 192, 0, 0);
  __builtin_prefetch(a1p + 192, 0, 0);
  __builtin_prefetch(b0p + 192, 0, 0);
  __builtin_prefetch(b1p + 192, 0, 0);
  v8f acc00 = {}, acc01 = {}, acc10 = {}, acc11 = {};
  for (int ks = 0; ks < K; ks += 32) {
    v16bf aF0 = afrag(a0p + ks);
    v16bf aF1 = afrag(a1p + ks);
    v16bf bF0 = afrag(b0p + ks);
    v16bf bF1 = afrag(b1p + ks);
    acc00 = wmma_bf(aF0, bF0, acc00);
    acc01 = wmma_bf(aF0, bF1, acc01);
    acc10 = wmma_bf(aF1, bF0, acc10);
    acc11 = wmma_bf(aF1, bF1, acc11);
  }
  const int rsub = (lane & 16) ? 8 : 0;
  const int csub = lane & 15;
  #pragma unroll
  for (int e = 0; e < 8; ++e) {
    size_t r0 = m0 + e + rsub, r1 = r0 + 16;
    size_t cA = n0 + csub,     cB = cA + 16;
    Cc[r0*Ncols + cA] = (__bf16)acc00[e];
    Cc[r0*Ncols + cB] = (__bf16)acc01[e];
    Cc[r1*Ncols + cA] = (__bf16)acc10[e];
    Cc[r1*Ncols + cB] = (__bf16)acc11[e];
  }
}

// ---------------- two-pass flash agent attention: one block per (b,h) ----------------
// agent_v[b,h,a,d] = softmax_n(SCALE*ah.khT + pos_bias) @ vh
__global__ __launch_bounds__(256) void flash_agent_kernel(const __bf16* __restrict__ agent_bf,
                                                          const __bf16* __restrict__ kv_bf,
                                                          const float*  __restrict__ pos_bias,
                                                          float* __restrict__ agent_v) {
  __shared__ __bf16 Aah[64*64];   // scaled agents, rows>=49 / cols>=48 zero
  __shared__ __bf16 Kc [64*64];   // K chunk [n_local][d]
  __shared__ float  Ssc[64*64];   // score chunk [a][n_local]
  __shared__ __bf16 Pp [64*64];   // probs (bf16) [a][n_local]
  __shared__ __bf16 Vc [64*64];   // V chunk transposed [d][n_local]
  __shared__ float  Oacc[64*48];
  __shared__ float  mrow[64];
  __shared__ float  lrow[64];

  const int tid  = threadIdx.x;
  const int lane = tid & 31;
  const int wave = tid >> 5;
  const int h = blockIdx.x & 7;
  const int b = blockIdx.x >> 3;

  for (int i = tid; i < 64*64; i += 256) {
    int a = i >> 6, d = i & 63;
    float val = 0.f;
    if (a < AG_ && d < DH_)
      val = (float)agent_bf[((size_t)(b*AG_ + a))*C_ + h*DH_ + d] * SCALE_;
    Aah[i] = (__bf16)val;
  }
  for (int i = tid; i < 64*48; i += 256) Oacc[i] = 0.f;
  if (tid < 64) { mrow[tid] = -1e30f; lrow[tid] = 0.f; }
  __syncthreads();

  const int aoff = (lane & 16) ? 8  : 0;
  const int boff = (lane & 16) ? 8  : 0;

  auto stageK = [&](int n0) {
    for (int i = tid; i < 64*64; i += 256) {
      int j = i >> 6, d = i & 63;
      __bf16 v = (__bf16)0.f;
      if (d < DH_) v = kv_bf[((size_t)(b*N_ + n0 + j))*(2*C_) + h*DH_ + d];
      Kc[i] = v;
    }
    // prefetch next chunk's K rows across the upcoming barrier-serialized compute
    int nn = n0 + 64;
    if (nn < N_ && tid < 64)
      __builtin_prefetch(&kv_bf[((size_t)(b*N_ + nn + tid))*(2*C_) + h*DH_], 0, 0);
  };
  auto computeS = [&]() {
    for (int t = wave; t < 16; t += 8) {   // 4 m-tiles x 4 n-tiles
      int mt = t >> 2, nt = t & 3;
      v8f acc = {};
      #pragma unroll
      for (int ks = 0; ks < 2; ++ks) {
        v16bf aF = afrag(&Aah[(mt*16 + (lane & 15))*64 + ks*32 + aoff]);
        v16bf bF = afrag(&Kc[(nt*16 + (lane & 15))*64 + ks*32 + boff]);
        acc = wmma_bf(aF, bF, acc);
      }
      int abase = mt*16 + ((lane & 16) ? 8 : 0);
      int j = nt*16 + (lane & 15);
      #pragma unroll
      for (int e = 0; e < 8; ++e) Ssc[(abase + e)*64 + j] = acc[e];
    }
  };

  // Pass A: row maxima
  for (int c = 0; c < N_/64; ++c) {
    int n0 = c*64;
    stageK(n0); __syncthreads();
    computeS(); __syncthreads();
    if (tid < AG_) {
      const float* bp = pos_bias + ((size_t)(h*AG_ + tid))*N_ + n0;
      float mx = mrow[tid];
      for (int j = 0; j < 64; ++j) mx = fmaxf(mx, Ssc[tid*64 + j] + bp[j]);
      mrow[tid] = mx;
    }
    __syncthreads();
  }

  // Pass B: exp / sum / P@V accumulate
  for (int c = 0; c < N_/64; ++c) {
    int n0 = c*64;
    stageK(n0); __syncthreads();
    computeS(); __syncthreads();
    if (tid < 64) {
      if (tid < AG_) {
        const float* bp = pos_bias + ((size_t)(h*AG_ + tid))*N_ + n0;
        float m = mrow[tid], s = 0.f;
        for (int j = 0; j < 64; ++j) {
          float e = __expf(Ssc[tid*64 + j] + bp[j] - m);
          Pp[tid*64 + j] = (__bf16)e;
          s += e;
        }
        lrow[tid] += s;
      } else {
        for (int j = 0; j < 64; ++j) Pp[tid*64 + j] = (__bf16)0.f;
      }
    }
    __syncthreads();
    for (int i = tid; i < 64*64; i += 256) {   // Vc[d][j]
      int d = i >> 6, j = i & 63;
      __bf16 v = (__bf16)0.f;
      if (d < DH_) v = kv_bf[((size_t)(b*N_ + n0 + j))*(2*C_) + C_ + h*DH_ + d];
      Vc[i] = v;
    }
    {   // prefetch next chunk's V rows
      int nn = n0 + 64;
      if (nn < N_ && tid < 64)
        __builtin_prefetch(&kv_bf[((size_t)(b*N_ + nn + tid))*(2*C_) + C_ + h*DH_], 0, 0);
    }
    __syncthreads();
    for (int t = wave; t < 12; t += 8) {       // 4 m-tiles x 3 d-tiles
      int mt = t / 3, nt = t % 3;
      v8f acc = {};
      #pragma unroll
      for (int ks = 0; ks < 2; ++ks) {
        v16bf aF = afrag(&Pp[(mt*16 + (lane & 15))*64 + ks*32 + aoff]);
        v16bf bF = afrag(&Vc[(nt*16 + (lane & 15))*64 + ks*32 + boff]);
        acc = wmma_bf(aF, bF, acc);
      }
      int abase = mt*16 + ((lane & 16) ? 8 : 0);
      int dcol  = nt*16 + (lane & 15);
      #pragma unroll
      for (int e = 0; e < 8; ++e) Oacc[(abase + e)*48 + dcol] += acc[e];
    }
    __syncthreads();
  }

  for (int i = tid; i < AG_*DH_; i += 256) {
    int a = i / DH_, d = i % DH_;
    agent_v[((size_t)((b*HEADS_ + h)*AG_ + a))*DH_ + d] = Oacc[a*48 + d] / lrow[a];
  }
}

// ---------------- token->agent attention, reduced directly to pooled partials ----------------
__global__ __launch_bounds__(128) void qattn_kernel(const __bf16* __restrict__ q_bf,
                                                    const __bf16* __restrict__ ag_bf,
                                                    const float*  __restrict__ agv,
                                                    const float*  __restrict__ agb,
                                                    float* __restrict__ ppart) {
  __shared__ float ah_s[AG_*DH_];
  __shared__ float av_s[AG_*DH_];
  __shared__ float lg[128*AG_];
  const int tid = threadIdx.x;
  const int ch  = blockIdx.x & 7;
  const int h   = (blockIdx.x >> 3) & 7;
  const int b   = blockIdx.x >> 6;
  for (int i = tid; i < AG_*DH_; i += 128) {
    int a = i / DH_, d = i % DH_;
    ah_s[i] = (float)ag_bf[((size_t)(b*AG_ + a))*C_ + h*DH_ + d];
    av_s[i] = agv[((size_t)((b*HEADS_ + h)*AG_ + a))*DH_ + d];
  }
  __syncthreads();
  float acc[DH_];
  #pragma unroll
  for (int d = 0; d < DH_; ++d) acc[d] = 0.f;
  float* myl = lg + tid*AG_;
  const int NCH = N_/8;   // 392
  for (int nl = tid; nl < NCH; nl += 128) {
    int n = ch*NCH + nl;
    float qv[DH_];
    const __bf16* qp = q_bf + ((size_t)(b*N_ + n))*C_ + h*DH_;
    if (nl + 128 < NCH)
      __builtin_prefetch(qp + (size_t)128*C_, 0, 0);
    #pragma unroll
    for (int d = 0; d < DH_; ++d) qv[d] = (float)qp[d] * SCALE_;
    const float* bp = agb + ((size_t)h*N_ + n)*AG_;
    float mx = -1e30f;
    for (int a = 0; a < AG_; ++a) {
      float s = bp[a];
      #pragma unroll
      for (int d = 0; d < DH_; ++d) s += qv[d]*ah_s[a*DH_ + d];
      myl[a] = s;
      mx = fmaxf(mx, s);
    }
    float ssum = 0.f;
    for (int a = 0; a < AG_; ++a) { float e = __expf(myl[a] - mx); myl[a] = e; ssum += e; }
    float inv = 1.f/ssum;
    for (int a = 0; a < AG_; ++a) {
      float wgt = myl[a]*inv;
      #pragma unroll
      for (int d = 0; d < DH_; ++d) acc[d] += wgt*av_s[a*DH_ + d];
    }
  }
  __syncthreads();                 // done with lg as logits; reuse for reduction
  #pragma unroll
  for (int d = 0; d < DH_; ++d) lg[tid*DH_ + d] = acc[d];
  __syncthreads();
  if (tid < DH_) {
    float s = 0.f;
    for (int t = 0; t < 128; ++t) s += lg[t*DH_ + tid];
    ppart[((size_t)((b*HEADS_ + h)*8 + ch))*DH_ + tid] = s;
  }
}

// ---------------- mean of depthwise 3x3 conv via inclusion-exclusion ----------------
__global__ void dwc_pool_kernel(const __bf16* __restrict__ kv_bf, const float* __restrict__ w,
                                const float* __restrict__ bias, float* __restrict__ out) {
  int idx = blockIdx.x*256 + threadIdx.x;
  if (idx >= B_*C_) return;
  int c = idx % C_, b = idx / C_;
  const __bf16* vp = kv_bf + (size_t)b*N_*(2*C_) + C_ + c;
  float T = 0, r0 = 0, r55 = 0, c0 = 0, c55 = 0;
  float v00 = 0, v0e = 0, ve0 = 0, vee = 0;
  for (int y = 0; y < H_; ++y)
    for (int x = 0; x < H_; ++x) {
      float val = (float)vp[(size_t)(y*H_ + x)*(2*C_)];
      T += val;
      if (y == 0)  r0  += val;
      if (y == 55) r55 += val;
      if (x == 0)  c0  += val;
      if (x == 55) c55 += val;
      if (y == 0  && x == 0)  v00 = val;
      if (y == 0  && x == 55) v0e = val;
      if (y == 55 && x == 0)  ve0 = val;
      if (y == 55 && x == 55) vee = val;
    }
  float s = 0.f;
  for (int ky = 0; ky < 3; ++ky) {
    int dy = ky - 1;
    for (int kx = 0; kx < 3; ++kx) {
      int dx = kx - 1;
      float R = T;
      if (dy == -1) R -= r55; else if (dy == 1) R -= r0;
      if (dx == -1) R -= c55; else if (dx == 1) R -= c0;
      if (dy != 0 && dx != 0)
        R += (dy == -1) ? ((dx == -1) ? vee : ve0) : ((dx == -1) ? v0e : v00);
      s += w[c*9 + ky*3 + kx] * R;
    }
  }
  out[idx] = bias[c] + s * (1.f/(float)N_);
}

// ---------------- combine pooled partials ----------------
__global__ void tail1_kernel(const float* __restrict__ ppart, const float* __restrict__ pdwc,
                             float* __restrict__ ppre) {
  int idx = blockIdx.x*256 + threadIdx.x;
  if (idx >= B_*C_) return;
  int c = idx % C_, b = idx / C_;
  int h = c / DH_, d = c % DH_;
  float s = 0.f;
  for (int ch = 0; ch < 8; ++ch)
    s += ppart[((size_t)((b*HEADS_ + h)*8 + ch))*DH_ + d];
  ppre[idx] = s/(float)N_ + pdwc[idx];
}

// ---------------- proj + LN + MLP head (tiny): one block per batch ----------------
__global__ __launch_bounds__(256) void tail2_kernel(const float* __restrict__ pooled,
    const float* __restrict__ Wp, const float* __restrict__ bp,
    const float* __restrict__ g,  const float* __restrict__ beta,
    const float* __restrict__ W1, const float* __restrict__ b1,
    const float* __restrict__ W2, const float* __restrict__ b2,
    float* __restrict__ out) {
  __shared__ float p[C_], z[C_], h1[2*C_], stat[2];
  int b = blockIdx.x, tid = threadIdx.x;
  for (int c = tid; c < C_; c += 256) p[c] = pooled[b*C_ + c];
  __syncthreads();
  for (int c = tid; c < C_; c += 256) {
    float acc = bp[c];
    for (int k = 0; k < C_; ++k) acc += p[k]*Wp[(size_t)k*C_ + c];
    z[c] = acc;
  }
  __syncthreads();
  if (tid == 0) {
    float m = 0.f; for (int c = 0; c < C_; ++c) m += z[c]; m /= C_;
    float v = 0.f; for (int c = 0; c < C_; ++c) { float d = z[c]-m; v += d*d; } v /= C_;
    stat[0] = m; stat[1] = rsqrtf(v + 1e-5f);
  }
  __syncthreads();
  for (int c = tid; c < C_; c += 256) z[c] = (z[c]-stat[0])*stat[1]*g[c] + beta[c];
  __syncthreads();
  for (int j = tid; j < 2*C_; j += 256) {
    float acc = b1[j];
    for (int k = 0; k < C_; ++k) acc += z[k]*W1[(size_t)k*(2*C_) + j];
    h1[j] = 0.5f*acc*(1.f + erff(acc*0.70710678118654752f));
  }
  __syncthreads();
  for (int o = tid; o < 1000; o += 256) {
    float acc = b2[o];
    for (int k = 0; k < 2*C_; ++k) acc += h1[k]*W2[(size_t)k*1000 + o];
    out[(size_t)b*1000 + o] = acc;
  }
}

// ---------------- launcher ----------------
extern "C" void kernel_launch(void* const* d_in, const int* in_sizes, int n_in,
                              void* d_out, int out_size, void* d_ws, size_t ws_size,
                              hipStream_t stream) {
  (void)in_sizes; (void)n_in; (void)out_size; (void)ws_size;
  const float* x     = (const float*)d_in[0];
  const float* Wq    = (const float*)d_in[1];
  const float* Wkv   = (const float*)d_in[2];
  const float* an_b  = (const float*)d_in[3];
  const float* na_b  = (const float*)d_in[4];
  const float* ah_b  = (const float*)d_in[5];
  const float* aw_b  = (const float*)d_in[6];
  const float* ha_b  = (const float*)d_in[7];
  const float* wa_b  = (const float*)d_in[8];
  const float* dwcw  = (const float*)d_in[9];
  const float* dwcb  = (const float*)d_in[10];
  const float* Wproj = (const float*)d_in[11];
  const float* bproj = (const float*)d_in[12];
  const float* lng   = (const float*)d_in[13];
  const float* lnb   = (const float*)d_in[14];
  const float* W1    = (const float*)d_in[15];
  const float* b1    = (const float*)d_in[16];
  const float* W2    = (const float*)d_in[17];
  const float* b2    = (const float*)d_in[18];
  float* out = (float*)d_out;

  char* wptr = (char*)d_ws;
  auto alloc = [&](size_t bytes) { char* p = wptr; wptr += (bytes + 255) & ~(size_t)255; return p; };
  __bf16* xbf   = (__bf16*)alloc((size_t)M_*C_*2);            // 38.5 MB
  __bf16* WqT   = (__bf16*)alloc((size_t)C_*C_*2);
  __bf16* WkvT  = (__bf16*)alloc((size_t)2*C_*C_*2);
  __bf16* q_bf  = (__bf16*)alloc((size_t)M_*C_*2);            // 38.5 MB
  __bf16* kv_bf = (__bf16*)alloc((size_t)M_*2*C_*2);          // 77 MB
  __bf16* ag_bf = (__bf16*)alloc((size_t)B_*AG_*C_*2);
  float*  posb  = (float*)alloc((size_t)HEADS_*AG_*N_*4);     // 4.9 MB
  float*  agb   = (float*)alloc((size_t)HEADS_*N_*AG_*4);     // 4.9 MB
  float*  agv   = (float*)alloc((size_t)B_*HEADS_*AG_*DH_*4);
  float*  ppart = (float*)alloc((size_t)B_*HEADS_*8*DH_*4);
  float*  pdwc  = (float*)alloc((size_t)B_*C_*4);
  float*  ppre  = (float*)alloc((size_t)B_*C_*4);

  // 1) convert / transpose / precompute biases
  { long n = (long)M_*C_;
    f32_to_bf16_kernel<<<(unsigned)((n + 255)/256), 256, 0, stream>>>(x, xbf, n); }
  transpose_bf16_kernel<<<(C_*C_ + 255)/256, 256, 0, stream>>>(Wq, WqT, C_);
  transpose_bf16_kernel<<<(2*C_*C_ + 255)/256, 256, 0, stream>>>(Wkv, WkvT, 2*C_);
  pos_bias_kernel<<<(HEADS_*AG_*N_ + 255)/256, 256, 0, stream>>>(an_b, ah_b, aw_b, posb);
  agent_bias_kernel<<<(HEADS_*N_*AG_ + 255)/256, 256, 0, stream>>>(na_b, ha_b, wa_b, agb);

  // 2) main WMMA GEMMs: q = x@Wq, kv = x@Wkv
  gemm_bf16_kernel<<<dim3(M_/128, C_/64),   256, 0, stream>>>(xbf, WqT,  q_bf,  C_);
  gemm_bf16_kernel<<<dim3(M_/128, 2*C_/64), 256, 0, stream>>>(xbf, WkvT, kv_bf, 2*C_);

  // 3) agent pooling
  agent_pool_kernel<<<(B_*AG_*C_ + 255)/256, 256, 0, stream>>>(q_bf, ag_bf);

  // 4) flash agent->token attention (WMMA)
  flash_agent_kernel<<<B_*HEADS_, 256, 0, stream>>>(ag_bf, kv_bf, posb, agv);

  // 5) token->agent attention reduced straight to pooled partials
  qattn_kernel<<<B_*HEADS_*8, 128, 0, stream>>>(q_bf, ag_bf, agv, agb, ppart);

  // 6) depthwise-conv mean (closed form)
  dwc_pool_kernel<<<(B_*C_ + 255)/256, 256, 0, stream>>>(kv_bf, dwcw, dwcb, pdwc);

  // 7) combine + head
  tail1_kernel<<<(B_*C_ + 255)/256, 256, 0, stream>>>(ppart, pdwc, ppre);
  tail2_kernel<<<B_, 256, 0, stream>>>(ppre, Wproj, bproj, lng, lnb, W1, b1, W2, b2, out);
}
